// PointCrossFeature_28475633173117
// MI455X (gfx1250) — matrix operand order repaired and checked by reading, not compiled, not run on hardware
//
#include <hip/hip_runtime.h>

typedef __attribute__((ext_vector_type(16))) _Float16 v16h;
typedef __attribute__((ext_vector_type(8)))  _Float16 v8h;
typedef __attribute__((ext_vector_type(8)))  float    v8f;
typedef __attribute__((ext_vector_type(2)))  float    v2f;

#define PCF_N   4096
#define BNI     0.999995f   // 1/sqrt(1+1e-5)

#if defined(__has_builtin)
#  if __has_builtin(__builtin_amdgcn_wmma_f32_16x16x4_f32)
#    define PCF_F32WMMA 1
#  endif
#endif
#ifndef PCF_F32WMMA
#  define PCF_F32WMMA 0
#endif

// dynamic LDS for the epilogue: 8 waves * (16x132) edge/fknn tiles + W1 (64x132)
#define PCF_EPI_LDSF (8 * 16 * 132 + 64 * 132)
#define PCF_EPI_LDSB (PCF_EPI_LDSF * 4)

__device__ __forceinline__ float pcf_leaky(float x) { return x > 0.f ? x : 0.2f * x; }
__device__ __forceinline__ float pcf_elu(float x)   { return x > 0.f ? x : (__expf(x) - 1.f); }

// ---------------------------------------------------------------------------
// Kernel 0: features [B,32,N] f32 -> row-major f16 [B,N,32] + f16-consistent |p|^2
// ---------------------------------------------------------------------------
__global__ __launch_bounds__(256) void pcf_prep(const float* __restrict__ feat,
                                                _Float16* __restrict__ ph,
                                                float* __restrict__ sq) {
  int t = blockIdx.x * 256 + threadIdx.x;      // t = b*N + n
  int b = t >> 12, n = t & (PCF_N - 1);
  const float* fb = feat + ((size_t)b << 17);  // b*32*4096
  float s = 0.f;
  #pragma unroll
  for (int c = 0; c < 32; ++c) {
    float x = fb[(c << 12) + n];
    _Float16 h = (_Float16)x;
    ph[((size_t)t << 5) + c] = h;
    float hf = (float)h;
    s += hf * hf;
  }
  sq[t] = s;
}

// ---------------------------------------------------------------------------
// Kernel 1: fused gram (WMMA f16->f32) + per-row top-16 selection.
// 4 waves/block, 16-row M-tile per wave, 2 column tiles per iteration
// (back-to-back WMMAs), double-buffered B fragments, L2-resident stream.
// ---------------------------------------------------------------------------
__global__ __launch_bounds__(128) void pcf_knn(const _Float16* __restrict__ ph,
                                               const float* __restrict__ sq,
                                               int* __restrict__ sel) {
  __shared__ float stile[4][2][16][17];
  __shared__ float lval[4][32][16];
  __shared__ int   lid [4][32][16];

  const int tid  = threadIdx.x;
  const int wave = tid >> 5;
  const int lane = tid & 31;
  const int lrow = lane & 15;
  const int hi   = lane >> 4;
  const int b    = blockIdx.y;
  const int rowbase = blockIdx.x * 64 + wave * 16;
  const _Float16* phb = ph + ((size_t)b * PCF_N << 5);
  const float*    sqb = sq + b * PCF_N;

  // A fragment: 16x32 f16, contiguous k = 16*hi + s (matches B)
  const _Float16* arow = phb + ((size_t)(rowbase + lrow) << 5);
  v8h alo = *(const v8h*)(arow + hi * 16);
  v8h ahi = *(const v8h*)(arow + hi * 16 + 8);
  v16h afrag;
  #pragma unroll
  for (int j = 0; j < 8; ++j) { afrag[j] = alo[j]; afrag[8 + j] = ahi[j]; }

  float sqr[8];
  #pragma unroll
  for (int r = 0; r < 8; ++r) sqr[r] = sqb[rowbase + r + 8 * hi];

  auto loadB = [&](int tile, v16h& bf, float& sqc) {
    const _Float16* brow = phb + ((size_t)(tile * 16 + lrow) << 5);
    bf  = *(const v16h*)(brow + hi * 16);
    sqc = sqb[tile * 16 + lrow];
  };

  float tv[16]; int tix[16];
  #pragma unroll
  for (int j = 0; j < 16; ++j) { tv[j] = -3e38f; tix[j] = 0; }
  float minv = -3e38f; int minslot = 0;

  v16h b0, b1; float q0, q1;
  loadB(0, b0, q0);
  loadB(1, b1, q1);

  const int NIT = PCF_N / 32;
  for (int it = 0; it < NIT; ++it) {
    const int nxt = ((it + 1) & (NIT - 1)) * 2;
    v16h nb0, nb1; float nq0, nq1;
    loadB(nxt,     nb0, nq0);
    loadB(nxt + 1, nb1, nq1);
    __builtin_prefetch(phb + ((size_t)((((it * 2 + 8) & 255) * 16) + lrow) << 5), 0, 3);

    v8f c0 = {}, c1 = {};
    c0 = __builtin_amdgcn_wmma_f32_16x16x32_f16(false, afrag, false, b0,
                                                (short)0, c0, false, false);
    c1 = __builtin_amdgcn_wmma_f32_16x16x32_f16(false, afrag, false, b1,
                                                (short)0, c1, false, false);

    #pragma unroll
    for (int r = 0; r < 8; ++r) {
      stile[wave][0][r + 8 * hi][lrow] = 2.0f * c0[r] - sqr[r] - q0;
      stile[wave][1][r + 8 * hi][lrow] = 2.0f * c1[r] - sqr[r] - q1;
    }
    __syncthreads();

    const int colbase = it * 32;
    #pragma unroll
    for (int u = 0; u < 2; ++u) {
      #pragma unroll
      for (int j = 0; j < 8; ++j) {
        float s = stile[wave][u][lrow][hi * 8 + j];
        int   id = colbase + u * 16 + hi * 8 + j;
        if (s > minv) {
          #pragma unroll
          for (int q = 0; q < 16; ++q)
            if (q == minslot) { tv[q] = s; tix[q] = id; }
          minv = tv[0]; minslot = 0;
          #pragma unroll
          for (int q = 1; q < 16; ++q)
            if (tv[q] < minv) { minv = tv[q]; minslot = q; }
        }
      }
    }
    __syncthreads();

    b0 = nb0; b1 = nb1; q0 = nq0; q1 = nq1;
  }

  #pragma unroll
  for (int j = 0; j < 16; ++j) { lval[wave][lane][j] = tv[j]; lid[wave][lane][j] = tix[j]; }
  __syncthreads();

  if (lane < 16) {
    const int row = rowbase + lane;
    float cv[32]; int ci[32];
    #pragma unroll
    for (int j = 0; j < 16; ++j) {
      cv[j]      = lval[wave][lane][j];      ci[j]      = lid[wave][lane][j];
      cv[16 + j] = lval[wave][lane + 16][j]; ci[16 + j] = lid[wave][lane + 16][j];
    }
    for (int o = 0; o < 16; ++o) {
      int best = 0; float bv = cv[0];
      for (int q = 1; q < 32; ++q)
        if (cv[q] > bv) { bv = cv[q]; best = q; }
      sel[((size_t)(b * PCF_N + row) << 4) + o] = ci[best];
      cv[best] = -3e38f;
    }
  }
}

// ---------------------------------------------------------------------------
// Kernel 2: fused epilogue — one thread per (b,n,k).
// conv1 (64x128, 2/3 of the epilogue MACs) runs on the f32-exact matrix pipe
// (v_wmma_f32_16x16x4_f32) when the builtin exists: per wave, 16 pairs form an
// N-tile; edge columns staged in LDS (stride 132 -> conflict-free), W1 staged
// once per block; 32 K-chunks x 4 M-tiles = 128 WMMAs per subtile.
// ---------------------------------------------------------------------------
__global__ __launch_bounds__(256) void pcf_epilogue(
    const float* __restrict__ coords, const float* __restrict__ feat,
    const int* __restrict__ kidx, const float* __restrict__ kdist,
    const float* __restrict__ W1, const float* __restrict__ W2,
    const float* __restrict__ W3, const float* __restrict__ W4,
    const float* __restrict__ W5,
    const float* __restrict__ g1, const float* __restrict__ b1,
    const float* __restrict__ g2, const float* __restrict__ b2,
    const float* __restrict__ g3, const float* __restrict__ b3,
    const float* __restrict__ g4, const float* __restrict__ b4,
    const float* __restrict__ g5, const float* __restrict__ b5,
    const int* __restrict__ sel, float* __restrict__ out) {
  const int t = blockIdx.x * 256 + threadIdx.x;  // b*N*K + n*K + k
  const int k = t & 15, n = (t >> 4) & (PCF_N - 1), b = t >> 16;

  const int s = sel[(size_t)t];
  const float* fb = feat + ((size_t)b << 17);

  float ctr[32], nbr[32];
  #pragma unroll
  for (int c = 0; c < 32; ++c) {
    ctr[c] = fb[(c << 12) + n];
    nbr[c] = fb[(c << 12) + s];
  }
  float dot = 0.f, sc = 0.f, sn = 0.f;
  #pragma unroll
  for (int c = 0; c < 32; ++c) {
    dot += ctr[c] * nbr[c]; sc += ctr[c] * ctr[c]; sn += nbr[c] * nbr[c];
  }
  const float dist = 2.f * dot - sc - sn;        // exact f32 neg-sqdist

  float fknn[64];

#if PCF_F32WMMA
  extern __shared__ float smem[];
  float* W1ls = smem + 8 * 16 * 132;
  {  // stage W1 into LDS once per block (row stride 132 -> 16 distinct bank groups)
    for (int idx = threadIdx.x; idx < 64 * 128; idx += 256)
      W1ls[(idx >> 7) * 132 + (idx & 127)] = W1[idx];
  }
  __syncthreads();

  const int wv = threadIdx.x >> 5;
  const int ln = threadIdx.x & 31;
  const int lr = ln & 15;
  const int hh = ln >> 4;
  float* Ew = smem + wv * 16 * 132;              // per-wave 16x132 tile

  #pragma unroll
  for (int sub = 0; sub < 2; ++sub) {            // two 16-pair subtiles per wave
    if (hh == sub) {                             // owners stage their edge column
      float* col = Ew + lr * 132;
      #pragma unroll
      for (int c = 0; c < 32; ++c) {
        col[c]      = ctr[c];
        col[32 + c] = nbr[c];
        col[64 + c] = nbr[c] - ctr[c];
        col[96 + c] = dist;
      }
    }
    __syncthreads();

    // FKNN[64x16] = W1[64x128] * E[128x16] on the f32 matrix pipe.
    // A (16x4 f32): lane m row, VGPR0/1 = K = 2*hh {+0,+1}.  B (4x16): symmetric.
    v8f acc[4];
    #pragma unroll
    for (int mt = 0; mt < 4; ++mt) acc[mt] = (v8f){};
    const float* Ecol = Ew + lr * 132;
    for (int kc = 0; kc < 32; ++kc) {
      v2f bf;
      bf[0] = Ecol[kc * 4 + 2 * hh];
      bf[1] = Ecol[kc * 4 + 2 * hh + 1];
      #pragma unroll
      for (int mt = 0; mt < 4; ++mt) {
        const float* wr = W1ls + (mt * 16 + lr) * 132 + kc * 4 + 2 * hh;
        v2f af; af[0] = wr[0]; af[1] = wr[1];
        acc[mt] = __builtin_amdgcn_wmma_f32_16x16x4_f32(false, af, false, bf,
                                                        (short)0, acc[mt], false, false);
      }
    }
    __syncthreads();

    // bn1 + leaky on C fragments, store FKNN back into the tile ([pair][ch]).
    #pragma unroll
    for (int mt = 0; mt < 4; ++mt) {
      #pragma unroll
      for (int r = 0; r < 8; ++r) {
        const int m = mt * 16 + r + 8 * hh;
        Ew[lr * 132 + m] = pcf_leaky(g1[m] * (acc[mt][r] * BNI) + b1[m]);
      }
    }
    __syncthreads();

    if (hh == sub) {                             // owner pulls its 64-ch column back
      #pragma unroll
      for (int c = 0; c < 64; ++c) fknn[c] = Ew[lr * 132 + c];
    }
    __syncthreads();
  }
#else
  // Fallback: scalar conv1.
  #pragma unroll 8
  for (int o = 0; o < 64; ++o) {
    const float* w = W1 + (o << 7);
    float a = 0.f;
    #pragma unroll
    for (int c = 0; c < 32; ++c) {
      float d = nbr[c] - ctr[c];
      a += w[c] * ctr[c] + w[32 + c] * nbr[c] + w[64 + c] * d + w[96 + c] * dist;
    }
    fknn[o] = pcf_leaky(g1[o] * (a * BNI) + b1[o]);
  }
#endif

  // ---- common tail: W2/W5(fknn part) accumulate from fknn ----
  float acc2[3] = {0.f, 0.f, 0.f};
  float acc5[32];
  #pragma unroll
  for (int i = 0; i < 32; ++i) acc5[i] = 0.f;
  #pragma unroll
  for (int o = 0; o < 64; ++o) {
    const float fk = fknn[o];
    #pragma unroll
    for (int j = 0; j < 3; ++j) acc2[j] += W2[(j << 6) + o] * fk;
    #pragma unroll
    for (int i = 0; i < 32; ++i) acc5[i] += W5[i * 96 + o] * fk;
  }

  // concat_point (13): [ctr3, nbr3, ctr3-nbr3, knn_dist, offset+ctr3]
  const int nid = kidx[(size_t)t];
  float cp[13];
  #pragma unroll
  for (int j = 0; j < 3; ++j) {
    const float c3  = coords[(size_t)(b * PCF_N + n)   * 3 + j];
    const float nb3 = coords[(size_t)(b * PCF_N + nid) * 3 + j];
    const float off = pcf_elu(g2[j] * (acc2[j] * BNI) + b2[j]);
    cp[j] = c3; cp[3 + j] = nb3; cp[6 + j] = c3 - nb3; cp[10 + j] = off + c3;
  }
  cp[9] = kdist[(size_t)t];

  // features_e = elu(bn3(W3*cp)) + ctr ; fold its W5 contribution in.
  #pragma unroll
  for (int c = 0; c < 32; ++c) {
    const float* w = W3 + c * 13;
    float a3 = 0.f;
    #pragma unroll
    for (int j = 0; j < 13; ++j) a3 += w[j] * cp[j];
    const float fe = pcf_elu(g3[c] * (a3 * BNI) + b3[c]) + ctr[c];
    #pragma unroll
    for (int i = 0; i < 32; ++i) acc5[i] += W5[i * 96 + 64 + c] * fe;
  }

  // out = [elu(bn4(W4*cp)) ; elu(bn5(acc5))], channel-major, coalesced over k.
  #pragma unroll
  for (int i = 0; i < 32; ++i) {
    const float* w = W4 + i * 13;
    float a4 = 0.f;
    #pragma unroll
    for (int j = 0; j < 13; ++j) a4 += w[j] * cp[j];
    const float op  = pcf_elu(g4[i] * (a4 * BNI) + b4[i]);
    const float ofv = pcf_elu(g5[i] * (acc5[i] * BNI) + b5[i]);
    out[(((size_t)(b * 64 + i)      * PCF_N) + n) * 16 + k] = op;
    out[(((size_t)(b * 64 + 32 + i) * PCF_N) + n) * 16 + k] = ofv;
  }
}

// ---------------------------------------------------------------------------
extern "C" void kernel_launch(void* const* d_in, const int* in_sizes, int n_in,
                              void* d_out, int out_size, void* d_ws, size_t ws_size,
                              hipStream_t stream) {
  (void)in_sizes; (void)n_in; (void)out_size; (void)ws_size;
  const float* coords = (const float*)d_in[0];
  const float* feat   = (const float*)d_in[1];
  const int*   kidx   = (const int*)d_in[2];
  const float* kdist  = (const float*)d_in[3];
  const float* W1 = (const float*)d_in[4];
  const float* W2 = (const float*)d_in[5];
  const float* W3 = (const float*)d_in[6];
  const float* W4 = (const float*)d_in[7];
  const float* W5 = (const float*)d_in[8];
  const float* g1 = (const float*)d_in[9],  * b1 = (const float*)d_in[10];
  const float* g2 = (const float*)d_in[11], * b2 = (const float*)d_in[12];
  const float* g3 = (const float*)d_in[13], * b3 = (const float*)d_in[14];
  const float* g4 = (const float*)d_in[15], * b4 = (const float*)d_in[16];
  const float* g5 = (const float*)d_in[17], * b5 = (const float*)d_in[18];

  char* ws = (char*)d_ws;
  _Float16* ph  = (_Float16*)ws;                            // 1 MB
  float*    sq  = (float*)(ws + (1 << 20));                 // 64 KB
  int*      sel = (int*)(ws + (1 << 20) + (1 << 16));       // 1 MB

  pcf_prep<<<64, 256, 0, stream>>>(feat, ph, sq);
  dim3 gknn(PCF_N / 64, 4);
  pcf_knn<<<gknn, 128, 0, stream>>>(ph, sq, sel);
  // dynamic-LDS size passed unconditionally (fallback path simply ignores it)
  pcf_epilogue<<<1024, 256, PCF_EPI_LDSB, stream>>>(coords, feat, kidx, kdist,
                                         W1, W2, W3, W4, W5,
                                         g1, b1, g2, b2, g3, b3, g4, b4, g5, b5,
                                         sel, (float*)d_out);
}
